// PaiNN_63067299774961
// MI455X (gfx1250) — matrix operand order repaired and checked by reading, not compiled, not run on hardware
//
#include <hip/hip_runtime.h>
#include <math.h>

#define N_ATOMS 10000
#define E_EDGES 320000
#define HID 128
#define NRBF 20
#define NLAY 2
#define NBATCH 100
#define FCUT 5.0f
#define FEPS 1e-8f

typedef __attribute__((ext_vector_type(16))) _Float16 v16h;
typedef __attribute__((ext_vector_type(8)))  _Float16 v8h;
typedef __attribute__((ext_vector_type(8)))  float    v8f;

__device__ __forceinline__ float silu_fast(float v) {
  return v * __builtin_amdgcn_rcpf(1.f + __expf(-v));  // v * sigmoid(v), v_rcp_f32
}

// ---------------------------------------------------------------------------
// WMMA GEMM: C[M x Nc] = act(A[M x K] @ B[K x Nc] + bias)
// Block = 128 threads (4 waves) -> 64x64 output tile.
// A staged row-major (vector ds_load_b128 per 8-f16 run); B staged
// FRAGMENT-MAJOR: sBf[tile][lane][16 f16] so each lane reads its whole
// fragment with 2x ds_load_b128. 4 v_wmma accumulations per wave per K-step.
// Requires K % 32 == 0, Nc % 64 == 0, M % 16 == 0 (all true here).
// ---------------------------------------------------------------------------
__global__ void __launch_bounds__(128) gemm_wmma_f16(
    const float* __restrict__ A, const float* __restrict__ Bm,
    const float* __restrict__ bias, float* __restrict__ C,
    int M, int K, int Nc, int act) {
  __shared__ _Float16 sA[64][40];            // rows 80B apart; 16B-aligned runs
  __shared__ _Float16 sBf[4 * 32 * 16];      // [tile][lane][k-frag], 4KB

  const int tid  = threadIdx.x;
  const int lane = tid & 31;
  const int wv   = tid >> 5;
  const int mblk = blockIdx.x * 64;
  const int n0   = blockIdx.y * 64;
  const int m0   = mblk + wv * 16;
  const int half = lane >> 4;
  const int l15  = lane & 15;

  // B staging geometry (constant over K): thread covers k-pair row rp (0..15),
  // cols cb..cb+7.  Element (r=2rp(+1), c) lives at fragment
  // [t=c>>4][laneB=(r>>4)*16+(c&15)][i=r&15]; the k-pair packs one u32.
  const int rp = tid >> 3;              // (tid*8)/64
  const int cb = (tid & 7) * 8;
  const int iB = rp & 7;                // u32 slot = i/2
  const int laneB_base = (rp >> 3) * 16;

  v8f acc0 = {0,0,0,0,0,0,0,0};
  v8f acc1 = {0,0,0,0,0,0,0,0};
  v8f acc2 = {0,0,0,0,0,0,0,0};
  v8f acc3 = {0,0,0,0,0,0,0,0};

  for (int k0 = 0; k0 < K; k0 += 32) {
    // ---- stage A tile 64x32: thread -> row tid/2, 16-col run ----
    {
      const int r = tid >> 1;
      const int c = (tid & 1) * 16;
      int row = mblk + r; if (row >= M) row = M - 1;
      const float* src = A + (size_t)row * K + k0 + c;
      v8h h0, h1;
      #pragma unroll
      for (int j = 0; j < 8; ++j) { h0[j] = (_Float16)src[j]; h1[j] = (_Float16)src[8 + j]; }
      *(v8h*)&sA[r][c]     = h0;
      *(v8h*)&sA[r][c + 8] = h1;
    }
    // ---- stage B tile 32x64 in fragment-major layout (8 packed pairs) ----
    {
      const float* r0 = Bm + (size_t)(k0 + 2 * rp) * Nc + n0 + cb;
      const float* r1 = r0 + Nc;
      unsigned int* sBu = (unsigned int*)sBf;
      #pragma unroll
      for (int j = 0; j < 8; ++j) {
        const int c  = cb + j;
        const int t  = c >> 4;
        const int ln = laneB_base + (c & 15);
        union { _Float16 h[2]; unsigned int u; } pk;
        pk.h[0] = (_Float16)r0[j];
        pk.h[1] = (_Float16)r1[j];
        sBu[(t * 32 + ln) * 8 + iB] = pk.u;
      }
    }
    if (k0 + 32 < K) {
      int prow = mblk + (tid >> 1); if (prow >= M) prow = M - 1;
      __builtin_prefetch(A + (size_t)prow * K + k0 + 32, 0, 1);
      __builtin_prefetch(Bm + (size_t)(k0 + 32 + (tid & 31)) * Nc + n0, 0, 1);
    }
    __syncthreads();

    // A fragment: two contiguous 8-f16 runs -> 2x ds_load_b128
    const int ar = wv * 16 + l15;
    const v8h alo = *(const v8h*)&sA[ar][half * 16];
    const v8h ahi = *(const v8h*)&sA[ar][half * 16 + 8];
    const v16h a = __builtin_shufflevector(alo, ahi,
        0, 1, 2, 3, 4, 5, 6, 7, 8, 9, 10, 11, 12, 13, 14, 15);

    // B fragments: contiguous 32B per lane -> 2x ds_load_b128 each
    v16h b0, b1, b2, b3;
    {
      const v8h lo = *(const v8h*)&sBf[(0 * 32 + lane) * 16];
      const v8h hi = *(const v8h*)&sBf[(0 * 32 + lane) * 16 + 8];
      b0 = __builtin_shufflevector(lo, hi, 0,1,2,3,4,5,6,7,8,9,10,11,12,13,14,15);
    }
    {
      const v8h lo = *(const v8h*)&sBf[(1 * 32 + lane) * 16];
      const v8h hi = *(const v8h*)&sBf[(1 * 32 + lane) * 16 + 8];
      b1 = __builtin_shufflevector(lo, hi, 0,1,2,3,4,5,6,7,8,9,10,11,12,13,14,15);
    }
    {
      const v8h lo = *(const v8h*)&sBf[(2 * 32 + lane) * 16];
      const v8h hi = *(const v8h*)&sBf[(2 * 32 + lane) * 16 + 8];
      b2 = __builtin_shufflevector(lo, hi, 0,1,2,3,4,5,6,7,8,9,10,11,12,13,14,15);
    }
    {
      const v8h lo = *(const v8h*)&sBf[(3 * 32 + lane) * 16];
      const v8h hi = *(const v8h*)&sBf[(3 * 32 + lane) * 16 + 8];
      b3 = __builtin_shufflevector(lo, hi, 0,1,2,3,4,5,6,7,8,9,10,11,12,13,14,15);
    }

    acc0 = __builtin_amdgcn_wmma_f32_16x16x32_f16(false, a, false, b0, (short)0, acc0, false, false);
    acc1 = __builtin_amdgcn_wmma_f32_16x16x32_f16(false, a, false, b1, (short)0, acc1, false, false);
    acc2 = __builtin_amdgcn_wmma_f32_16x16x32_f16(false, a, false, b2, (short)0, acc2, false, false);
    acc3 = __builtin_amdgcn_wmma_f32_16x16x32_f16(false, a, false, b3, (short)0, acc3, false, false);
    __syncthreads();
  }

  if (m0 < M) {                                   // wave-uniform, after all barriers
    const float bv0 = bias ? bias[n0 + l15]      : 0.f;
    const float bv1 = bias ? bias[n0 + 16 + l15] : 0.f;
    const float bv2 = bias ? bias[n0 + 32 + l15] : 0.f;
    const float bv3 = bias ? bias[n0 + 48 + l15] : 0.f;
    #pragma unroll
    for (int j = 0; j < 8; ++j) {
      const int row = m0 + half * 8 + j;          // C/D layout: VGPR j -> M = j + 8*half
      float* Crow = C + (size_t)row * Nc + n0;
      float v0 = acc0[j] + bv0, v1 = acc1[j] + bv1;
      float v2 = acc2[j] + bv2, v3 = acc3[j] + bv3;
      if (act) { v0 = silu_fast(v0); v1 = silu_fast(v1); v2 = silu_fast(v2); v3 = silu_fast(v3); }
      Crow[l15]      = v0;
      Crow[16 + l15] = v1;
      Crow[32 + l15] = v2;
      Crow[48 + l15] = v3;
    }
  }
}

// ---------------------------------------------------------------------------
__global__ void init_nodes(const float* __restrict__ emb, const int* __restrict__ atype,
                           float* __restrict__ q, float* __restrict__ mu) {
  const int i = blockIdx.x * blockDim.x + threadIdx.x;
  if (i < N_ATOMS * HID) {
    const int n = i >> 7, h = i & 127;
    q[i] = emb[atype[n] * HID + h];
  }
  if (i < N_ATOMS * 3 * HID) mu[i] = 0.f;
}

// ---------------------------------------------------------------------------
__global__ void edge_geom(const float* __restrict__ pos, const int* __restrict__ ei,
                          float* __restrict__ geom) {
  const int e = blockIdx.x * blockDim.x + threadIdx.x;
  if (e >= E_EDGES) return;
  const int s = ei[e], d = ei[E_EDGES + e];
  const float rx = pos[d * 3 + 0] - pos[s * 3 + 0];
  const float ry = pos[d * 3 + 1] - pos[s * 3 + 1];
  const float rz = pos[d * 3 + 2] - pos[s * 3 + 2];
  const float dist = sqrtf(rx * rx + ry * ry + rz * rz);
  const float inv = __builtin_amdgcn_rcpf(dist);
  const float Cc = (dist < FCUT)
      ? 0.5f * (__cosf(3.14159265358979f * dist / FCUT) + 1.f) : 0.f;
  float* g = geom + (size_t)e * 5;
  g[0] = rx * inv; g[1] = ry * inv; g[2] = rz * inv; g[3] = dist; g[4] = Cc;
}

// ---------------------------------------------------------------------------
// Fused edge message + scatter (L2-resident gathers + global f32 atomics).
// ---------------------------------------------------------------------------
__global__ void __launch_bounds__(128) edge_msg(
    const float* __restrict__ geom, const int* __restrict__ ei,
    const float* __restrict__ x, const float* __restrict__ mu_in,
    const float* __restrict__ filtW, const float* __restrict__ filtb,
    float* __restrict__ q, float* __restrict__ mu_out) {
  __shared__ float sW[NRBF * 3 * HID];   // 30 KB
  __shared__ float sB[3 * HID];
  __shared__ float sRbf[32][NRBF];
  __shared__ float sGeo[32][5];
  __shared__ int   sSrc[32], sDst[32];

  const int tid = threadIdx.x;
  for (int i = tid; i < NRBF * 3 * HID; i += 128) sW[i] = filtW[i];
  for (int i = tid; i < 3 * HID; i += 128)        sB[i] = filtb[i];

  const int e0 = blockIdx.x * 32;
  if (tid < 32) {
    const int e = e0 + tid;
    sSrc[tid] = ei[e];
    sDst[tid] = ei[E_EDGES + e];
    #pragma unroll
    for (int k = 0; k < 5; ++k) sGeo[tid][k] = geom[(size_t)e * 5 + k];
  }
  __syncthreads();

  for (int i = tid; i < 32 * NRBF; i += 128) {
    const int el = i / NRBF, r = i % NRBF;
    const float d = sGeo[el][3];
    const float c = (FCUT * r) / (NRBF - 1);
    const float t = d - c;
    sRbf[el][r] = __expf(-16.f * t * t);   // gamma = (R/CUTOFF)^2 = 16
  }
  __syncthreads();

  const int h = tid;  // 0..127
  for (int el = 0; el < 32; ++el) {
    const float Cc = sGeo[el][4];
    if (Cc == 0.f) continue;               // uniform branch: zero message
    const int sN = sSrc[el], dN = sDst[el];

    float w0 = sB[h], w1 = sB[HID + h], w2 = sB[2 * HID + h];
    #pragma unroll
    for (int r = 0; r < NRBF; ++r) {
      const float rb = sRbf[el][r];
      const float* wr = &sW[r * 3 * HID];
      w0 += rb * wr[h];
      w1 += rb * wr[HID + h];
      w2 += rb * wr[2 * HID + h];
    }
    const float* xs = x + (size_t)sN * (3 * HID);
    const float dq = Cc * w0 * xs[h];
    const float dR = Cc * w1 * xs[HID + h];
    const float dM = Cc * w2 * xs[2 * HID + h];

    const float* ms = mu_in + (size_t)sN * (3 * HID);
    atomicAdd(&q[(size_t)dN * HID + h], dq);
    float* md = mu_out + (size_t)dN * (3 * HID);
    atomicAdd(&md[h],           dR * sGeo[el][0] + dM * ms[h]);
    atomicAdd(&md[HID + h],     dR * sGeo[el][1] + dM * ms[HID + h]);
    atomicAdd(&md[2 * HID + h], dR * sGeo[el][2] + dM * ms[2 * HID + h]);
  }
}

// ---------------------------------------------------------------------------
__global__ void build_ctx(const float* __restrict__ q, const float* __restrict__ mumix,
                          float* __restrict__ ctx) {
  const int i = blockIdx.x * blockDim.x + threadIdx.x;
  if (i >= N_ATOMS * HID) return;
  const int n = i >> 7, h = i & 127;
  const float* mm = mumix + (size_t)n * 768;
  const float v0 = mm[h], v1 = mm[256 + h], v2 = mm[512 + h];
  const float vn = sqrtf(v0 * v0 + v1 * v1 + v2 * v2 + FEPS);
  ctx[(size_t)n * 256 + h]       = q[i];
  ctx[(size_t)n * 256 + 128 + h] = vn;
}

// ---------------------------------------------------------------------------
__global__ void mix_update(const float* __restrict__ mumix, const float* __restrict__ xi,
                           float* __restrict__ q, float* __restrict__ mu) {
  const int i = blockIdx.x * blockDim.x + threadIdx.x;
  if (i >= N_ATOMS * HID) return;
  const int n = i >> 7, h = i & 127;
  const float* mm = mumix + (size_t)n * 768;
  const float scal = mm[h] * mm[128 + h] + mm[256 + h] * mm[384 + h]
                   + mm[512 + h] * mm[640 + h];
  const float* xr = xi + (size_t)n * 384;
  q[i] += xr[h] + xr[256 + h] * scal;
  const float dmu = xr[128 + h];
  float* mr = mu + (size_t)n * 384;
  mr[h]       += dmu * mm[128 + h];
  mr[128 + h] += dmu * mm[384 + h];
  mr[256 + h] += dmu * mm[640 + h];
}

__global__ void zero_out(float* __restrict__ out) {
  const int i = blockIdx.x * blockDim.x + threadIdx.x;
  if (i < NBATCH) out[i] = 0.f;
}

__global__ void energy_reduce(const float* __restrict__ h2, const float* __restrict__ w2,
                              const float* __restrict__ b2, const int* __restrict__ batch,
                              float* __restrict__ out) {
  const int n = blockIdx.x * blockDim.x + threadIdx.x;
  if (n >= N_ATOMS) return;
  float acc = b2[0];
  const float* hr = h2 + (size_t)n * 64;
  #pragma unroll
  for (int j = 0; j < 64; ++j) acc += hr[j] * w2[j];
  atomicAdd(&out[batch[n]], acc);
}

// ---------------------------------------------------------------------------
extern "C" void kernel_launch(void* const* d_in, const int* in_sizes, int n_in,
                              void* d_out, int out_size, void* d_ws, size_t ws_size,
                              hipStream_t stream) {
  const float* pos    = (const float*)d_in[0];
  const int*   atype  = (const int*)d_in[1];
  const int*   ei     = (const int*)d_in[2];
  const int*   batch  = (const int*)d_in[3];
  const float* emb    = (const float*)d_in[4];
  const float* filtW  = (const float*)d_in[5];
  const float* filtb  = (const float*)d_in[6];
  const float* ctxW1  = (const float*)d_in[7];
  const float* ctxb1  = (const float*)d_in[8];
  const float* ctxW2  = (const float*)d_in[9];
  const float* ctxb2  = (const float*)d_in[10];
  const float* mixW   = (const float*)d_in[11];
  const float* mctxW1 = (const float*)d_in[12];
  const float* mctxb1 = (const float*)d_in[13];
  const float* mctxW2 = (const float*)d_in[14];
  const float* mctxb2 = (const float*)d_in[15];
  const float* outW1  = (const float*)d_in[16];
  const float* outb1  = (const float*)d_in[17];
  const float* outW2  = (const float*)d_in[18];
  const float* outb2  = (const float*)d_in[19];
  float* out = (float*)d_out;

  float* ws = (float*)d_ws;
  size_t off = 0;
  float* q     = ws + off; off += (size_t)N_ATOMS * HID;
  float* muA   = ws + off; off += (size_t)N_ATOMS * 3 * HID;
  float* muB   = ws + off; off += (size_t)N_ATOMS * 3 * HID;
  float* x     = ws + off; off += (size_t)N_ATOMS * 3 * HID;   // also reused for xi
  float* h1    = ws + off; off += (size_t)N_ATOMS * HID;       // also reused for out head
  float* mumix = ws + off; off += (size_t)N_ATOMS * 3 * 2 * HID;
  float* ctx   = ws + off; off += (size_t)N_ATOMS * 2 * HID;
  float* geom  = ws + off; off += (size_t)E_EDGES * 5;

  const int tot = N_ATOMS * 3 * HID;
  init_nodes<<<(tot + 127) / 128, 128, 0, stream>>>(emb, atype, q, muA);
  edge_geom<<<(E_EDGES + 127) / 128, 128, 0, stream>>>(pos, ei, geom);

  const int MG_N  = (N_ATOMS + 63) / 64;      // 157
  const int MG_3N = (3 * N_ATOMS + 63) / 64;  // 469

  float* mu_in = muA;
  float* mu_out = muB;
  for (int l = 0; l < NLAY; ++l) {
    gemm_wmma_f16<<<dim3(MG_N, HID / 64), 128, 0, stream>>>(
        q, ctxW1 + (size_t)l * HID * HID, ctxb1 + (size_t)l * HID, h1,
        N_ATOMS, HID, HID, 1);
    gemm_wmma_f16<<<dim3(MG_N, (3 * HID) / 64), 128, 0, stream>>>(
        h1, ctxW2 + (size_t)l * HID * 3 * HID, ctxb2 + (size_t)l * 3 * HID, x,
        N_ATOMS, HID, 3 * HID, 0);

    hipMemcpyAsync(mu_out, mu_in, (size_t)N_ATOMS * 3 * HID * sizeof(float),
                   hipMemcpyDeviceToDevice, stream);
    edge_msg<<<E_EDGES / 32, 128, 0, stream>>>(
        geom, ei, x, mu_in, filtW + (size_t)l * NRBF * 3 * HID,
        filtb + (size_t)l * 3 * HID, q, mu_out);

    gemm_wmma_f16<<<dim3(MG_3N, (2 * HID) / 64), 128, 0, stream>>>(
        mu_out, mixW + (size_t)l * HID * 2 * HID, nullptr, mumix,
        3 * N_ATOMS, HID, 2 * HID, 0);
    build_ctx<<<(N_ATOMS * HID + 127) / 128, 128, 0, stream>>>(q, mumix, ctx);
    gemm_wmma_f16<<<dim3(MG_N, (2 * HID) / 128), 128, 0, stream>>>(
        ctx, mctxW1 + (size_t)l * 2 * HID * HID, mctxb1 + (size_t)l * HID, h1,
        N_ATOMS, 2 * HID, HID, 1);
    gemm_wmma_f16<<<dim3(MG_N, (3 * HID) / 64), 128, 0, stream>>>(
        h1, mctxW2 + (size_t)l * HID * 3 * HID, mctxb2 + (size_t)l * 3 * HID, x,
        N_ATOMS, HID, 3 * HID, 0);
    mix_update<<<(N_ATOMS * HID + 127) / 128, 128, 0, stream>>>(mumix, x, q, mu_out);

    float* t = mu_in; mu_in = mu_out; mu_out = t;
  }

  gemm_wmma_f16<<<dim3(MG_N, 1), 128, 0, stream>>>(
      q, outW1, outb1, h1, N_ATOMS, HID, 64, 1);
  zero_out<<<1, 128, 0, stream>>>(out);
  energy_reduce<<<(N_ATOMS + 127) / 128, 128, 0, stream>>>(h1, outW2, outb2, batch, out);
}
